// LSTM_1735166788609
// MI455X (gfx1250) — matrix operand order repaired and checked by reading, not compiled, not run on hardware
//
#include <hip/hip_runtime.h>
#include <hip/hip_bf16.h>

// ---------------- CDNA5 WMMA types ----------------
typedef __attribute__((ext_vector_type(16))) _Float16 v16h;
typedef __attribute__((ext_vector_type(2)))  __fp16   v2hp;   // cvt_pkrtz result type
typedef __attribute__((ext_vector_type(8)))  float    v8f;

union Frag16 { v16h v; uint4 q[2]; };

#define T_STEPS 1024
#define BATCH   2048
#define HID     64
#define G4      256            // 4*HID gate columns
#define KAUG    96             // augmented K: 64 h + x0 + x1 + 1.0 + zero pad -> 3 x 32
#define WHS     96             // W-staging row stride in halves (192B, 16B aligned)
#define HS      104            // h-staging row stride in halves (208B; 52 dwords -> no bank dup)

// CDNA5 hardware tanh (TRANS class, co-executes with XDL WMMA)
#if __has_builtin(__builtin_amdgcn_tanhf)
__device__ __forceinline__ float htanh(float v) { return __builtin_amdgcn_tanhf(v); }
#else
__device__ __forceinline__ float htanh(float v) {
    float r; asm("v_tanh_f32 %0, %1" : "=v"(r) : "v"(v)); return r;
}
#endif
// sigmoid via hw tanh: sig(v) = 0.5*tanh(v/2) + 0.5  (mul + tanh + fma)
__device__ __forceinline__ float fsig(float v) {
    return __builtin_fmaf(0.5f, htanh(0.5f * v), 0.5f);
}

__global__ __launch_bounds__(256, 1)
void lstm_wmma_kernel(const float* __restrict__ x,
                      const float* __restrict__ W_ih,
                      const float* __restrict__ W_hh,
                      const float* __restrict__ b_ih,
                      const float* __restrict__ b_hh,
                      const float* __restrict__ W_fc,
                      const float* __restrict__ b_fc,
                      float* __restrict__ out) {
    // ---- LDS: 48K (W staging) + 13K (h ping-pong) + 0.5K (x async staging) = 61.5 KB ----
    __shared__ _Float16 lds_w[G4 * WHS];            // rows g: [W_hh[g,0:64] | wih0 | wih1 | bias | 0...]
    __shared__ _Float16 lds_h[2][2][16 * HS];       // [group][pingpong]: [h | x0 x1 1.0 | 0...]
    __shared__ float2   lds_xs[2][2][16];           // [group][pingpong] raw f32 x tiles (async dest)

    const int tid  = threadIdx.x;
    const int wave = tid >> 5;          // 0..7
    const int lane = tid & 31;
    const int n    = lane & 15;         // N col (C/D, B) == M row (A)
    const int kh   = lane >> 4;         // half-wave selector
    const int grp  = wave >> 2;         // 0..1 : 16-batch-row group within block
    const int jt   = wave & 3;          // hidden-column tile (16 cols) owned by this wave
    const int b0   = blockIdx.x * 32 + grp * 16;    // group's batch base
    const bool loader = (jt == 0);      // one wave per group does x feed and final FC

    // ---------------- phase 1: fill W staging + zero h buffers ----------------
    for (int idx = tid; idx < G4 * HID; idx += 256)
        lds_w[(idx >> 6) * WHS + (idx & 63)] = (_Float16)W_hh[idx];
    {   // augmented K rows: 64 -> W_ih col0, 65 -> W_ih col1, 66 -> fused bias, 67..95 -> 0
        const int g = tid;              // blockDim == 256 == G4
        lds_w[g * WHS + 64] = (_Float16)W_ih[2 * g + 0];
        lds_w[g * WHS + 65] = (_Float16)W_ih[2 * g + 1];
        lds_w[g * WHS + 66] = (_Float16)(b_ih[g] + b_hh[g]);
#pragma unroll
        for (int c = 67; c < KAUG; ++c) lds_w[g * WHS + c] = (_Float16)0.0f;
    }
    for (int idx = tid; idx < 2 * 2 * 16 * HS; idx += 256) {
        const int gb = idx / (16 * HS);
        ((_Float16*)lds_h[gb >> 1][gb & 1])[idx % (16 * HS)] = (_Float16)0.0f;
    }
    __syncthreads();

    // ---------------- phase 2: hoist B fragments; loader seeds x ----------------
    // 12 constant B fragments (4 gates x 3 K-steps) -> 96 VGPRs
    Frag16 Bf[4][3];
#pragma unroll
    for (int gg = 0; gg < 4; ++gg) {
        const _Float16* brow = lds_w + (16 * (jt + 4 * gg) + n) * WHS;
#pragma unroll
        for (int ks = 0; ks < 3; ++ks) {
            Bf[gg][ks].q[0] = *(const uint4*)(brow + 32 * ks + 8 * kh);
            Bf[gg][ks].q[1] = *(const uint4*)(brow + 32 * ks + 16 + 8 * kh);
        }
    }
    if (loader && lane < 16) {
        // "1.0" column (K=66) is constant in both ping-pong buffers
        lds_h[grp][0][lane * HS + 66] = (_Float16)1.0f;
        lds_h[grp][1][lane * HS + 66] = (_Float16)1.0f;
        // x[0] straight into buffer 0 cols 64/65 (f16)
        const float2 xv = ((const float2*)x)[b0 + lane];
        *(v2hp*)(lds_h[grp][0] + lane * HS + 64) = __builtin_amdgcn_cvt_pkrtz(xv.x, xv.y);
        // async-prefetch x[1] into staging (consumed during step 0)
        const unsigned ldsoff = (unsigned)(size_t)&lds_xs[grp][1][lane];
        const float2*  gp     = ((const float2*)x) + ((size_t)1 * BATCH + b0 + lane);
        asm volatile("global_load_async_to_lds_b64 %0, %1, off"
                     :: "v"(ldsoff), "v"(gp) : "memory");
    }
    __syncthreads();

    // cell state: c[r] for (batch row m = r + 8*kh, hidden col gi)
    const int gi = 16 * jt + n;
    float creg[8];
#pragma unroll
    for (int r = 0; r < 8; ++r) creg[r] = 0.0f;

    // ---------------- recurrence: gates = [h | x | 1] @ [W_hh ; W_ih ; b] ----------------
    for (int t = 0; t < T_STEPS; ++t) {
        const _Float16* hread  = lds_h[grp][t & 1];
        _Float16*       hwrite = lds_h[grp][(t + 1) & 1];

        // A fragments (16 x 96 f16, 3 K-steps), lane row m = n
        const _Float16* hrow = hread + n * HS;
        Frag16 A0, A1, A2;
        A0.q[0] = *(const uint4*)(hrow +  0 + 8 * kh);
        A0.q[1] = *(const uint4*)(hrow + 16 + 8 * kh);
        A1.q[0] = *(const uint4*)(hrow + 32 + 8 * kh);
        A1.q[1] = *(const uint4*)(hrow + 48 + 8 * kh);
        A2.q[0] = *(const uint4*)(hrow + 64 + 8 * kh);
        A2.q[1] = *(const uint4*)(hrow + 80 + 8 * kh);

        // loader: deliver x[t+1] into hwrite cols 64/65, then async-fetch x[t+2]
        if (loader) {
            if (t + 1 < T_STEPS) {
                asm volatile("s_wait_asynccnt 0x0" ::: "memory");
                if (lane < 16) {
                    const float2 xv = lds_xs[grp][(t + 1) & 1][lane];
                    *(v2hp*)(hwrite + lane * HS + 64) = __builtin_amdgcn_cvt_pkrtz(xv.x, xv.y);
                }
            }
            if (t + 2 < T_STEPS && lane < 16) {
                const unsigned ldsoff = (unsigned)(size_t)&lds_xs[grp][t & 1][lane];
                const float2*  gp     = ((const float2*)x) + ((size_t)(t + 2) * BATCH + b0 + lane);
                asm volatile("global_load_async_to_lds_b64 %0, %1, off"
                             :: "v"(ldsoff), "v"(gp) : "memory");
            }
        }

        // 12 WMMAs: 4 independent gate chains of 3 (K = 96)
        v8f acc[4];
#pragma unroll
        for (int gg = 0; gg < 4; ++gg) {
            v8f a = {};
            a = __builtin_amdgcn_wmma_f32_16x16x32_f16(false, A0.v, false, Bf[gg][0].v,
                                                       (short)0, a, false, false);
            a = __builtin_amdgcn_wmma_f32_16x16x32_f16(false, A1.v, false, Bf[gg][1].v,
                                                       (short)0, a, false, false);
            a = __builtin_amdgcn_wmma_f32_16x16x32_f16(false, A2.v, false, Bf[gg][2].v,
                                                       (short)0, a, false, false);
            acc[gg] = a;
        }

        // activations on hw v_tanh_f32 + state update (gx and bias already inside acc)
#pragma unroll
        for (int r = 0; r < 8; ++r) {
            const float ig = fsig (acc[0][r]);
            const float fg = fsig (acc[1][r]);
            const float gv = htanh(acc[2][r]);
            const float og = fsig (acc[3][r]);
            const float cv = __builtin_fmaf(fg, creg[r], ig * gv);
            creg[r] = cv;
            const float hv = og * htanh(cv);
            hwrite[(r + 8 * kh) * HS + gi] = (_Float16)hv;      // (m, j=gi)
        }

        __syncthreads();    // h ping-pong handoff among the 4 waves of the group
    }

    // ---------------- final FC: out[b] = h . W_fc + b_fc ----------------
    if (loader && lane < 16) {
        const _Float16* hr = lds_h[grp][T_STEPS & 1] + lane * HS;
        float s = 0.0f;
#pragma unroll
        for (int j = 0; j < HID; ++j)
            s = __builtin_fmaf((float)hr[j], W_fc[j], s);
        out[b0 + lane] = s + b_fc[0];
    }
}

extern "C" void kernel_launch(void* const* d_in, const int* in_sizes, int n_in,
                              void* d_out, int out_size, void* d_ws, size_t ws_size,
                              hipStream_t stream) {
    const float* x    = (const float*)d_in[0];
    const float* W_ih = (const float*)d_in[1];
    const float* W_hh = (const float*)d_in[2];
    const float* b_ih = (const float*)d_in[3];
    const float* b_hh = (const float*)d_in[4];
    const float* W_fc = (const float*)d_in[5];
    const float* b_fc = (const float*)d_in[6];
    float* out = (float*)d_out;

    // 64 blocks x 256 threads; 4 waves cooperate per 16-batch-row group (2 groups/block)
    lstm_wmma_kernel<<<BATCH / 32, 256, 0, stream>>>(x, W_ih, W_hh, b_ih, b_hh, W_fc, b_fc, out);
}
